// LinearAttention_369367187737
// MI455X (gfx1250) — compile-verified, matrix operand order checked by reading, not run on hardware
//
#include <hip/hip_runtime.h>
#include <hip/hip_bf16.h>
#include <math.h>

typedef __attribute__((ext_vector_type(16))) _Float16 v16h;
typedef __attribute__((ext_vector_type(8)))  _Float16 v8h;
typedef __attribute__((ext_vector_type(8)))  float    v8f;

#define B_     32
#define C_     512
#define P_     1024      // 32*32 pixels
#define HEADS_ 8
#define DH_    64
#define HID_   512       // HEADS_*DH_
#define NSEQ_  1028      // 4 mem slots + 1024 pixels
#define NPAD_  1056      // 33*32, zero padded
#define SQRT_C 22.62741699796952f

// ---------------- WMMA fragment helpers (wave32, 16x16x32 f16) ----------------
// A (16x32, MxK) and B (32x16, KxN as N-major rows of K): lane r=lane&15 owns
// row/col r; half hh=lane>>4 selects K chunks [8hh..8hh+7] and [16+8hh..23+8hh].
__device__ __forceinline__ v16h load_frag(const _Float16* __restrict__ base,
                                          int row0, int kk, int ld) {
  const int lane = threadIdx.x & 31;
  const int r  = lane & 15;
  const int hh = lane >> 4;
  const _Float16* p = base + (size_t)(row0 + r) * ld + kk + hh * 8;
  union { v16h v; v8h h[2]; } u;
  u.h[0] = *(const v8h*)(p);
  u.h[1] = *(const v8h*)(p + 16);
  return u.v;
}

__device__ __forceinline__ v8f wmma16(v16h a, v16h b, v8f c) {
  return __builtin_amdgcn_wmma_f32_16x16x32_f16(false, a, false, b, (short)0, c,
                                                false, false);
}

// D (16x16 f32): VGPR i -> row i + 8*(lane>>4), col = lane&15.
__device__ __forceinline__ void store_tile_f16(_Float16* base, size_t ld,
                                               int row0, int col0, v8f d) {
  const int lane = threadIdx.x & 31;
  const int r = lane & 15, hh = lane >> 4;
  _Float16* p = base + (size_t)(row0 + hh * 8) * ld + col0 + r;
#pragma unroll
  for (int i = 0; i < 8; ++i) p[(size_t)i * ld] = (_Float16)d[i];
}

__device__ __forceinline__ void store_tile_f32(float* base, size_t ld,
                                               int row0, int col0, v8f d) {
  const int lane = threadIdx.x & 31;
  const int r = lane & 15, hh = lane >> 4;
  float* p = base + (size_t)(row0 + hh * 8) * ld + col0 + r;
#pragma unroll
  for (int i = 0; i < 8; ++i) p[(size_t)i * ld] = d[i];
}

// transposed store: element (row,col) -> base[col*ldT + row] (8 contiguous halfs per lane)
__device__ __forceinline__ void store_tile_f16_T(_Float16* base, size_t ldT,
                                                 int row0, int col0, v8f d) {
  const int lane = threadIdx.x & 31;
  const int r = lane & 15, hh = lane >> 4;
  _Float16* p = base + (size_t)(col0 + r) * ldT + row0 + hh * 8;
#pragma unroll
  for (int i = 0; i < 8; ++i) p[i] = (_Float16)d[i];
}

// ---------------- small elementwise kernels ----------------
__global__ void convert_f16_kernel(const float* __restrict__ src,
                                   _Float16* __restrict__ dst, int n) {
  int i = blockIdx.x * blockDim.x + threadIdx.x;
  if (i < n) dst[i] = (_Float16)src[i];
}

// RMSNorm over channels + transpose to xn[b][p][c] (f16)
__global__ void rmsnorm_in_kernel(const float* __restrict__ x,
                                  const float* __restrict__ g,
                                  _Float16* __restrict__ xn) {
  int t = blockIdx.x * blockDim.x + threadIdx.x;   // b*1024+p, 32768 threads
  int b = t >> 10;
  int p = t & (P_ - 1);
  const float* xb = x + (size_t)b * C_ * P_ + p;
  float ss = 0.f;
  for (int c = 0; c < C_; ++c) { float v = xb[(size_t)c * P_]; ss += v * v; }
  float s = SQRT_C / fmaxf(sqrtf(ss), 1e-12f);
  _Float16* dst = xn + (size_t)t * C_;
  for (int c = 0; c < C_; ++c)
    dst[c] = (_Float16)(xb[(size_t)c * P_] * g[c] * s);
}

// ---------------- main WMMA GEMM: C[b] = A (MxK) * B[b]^T (NxK rows) ----------------
// Register double-buffered over the K dimension so each stage's global loads are
// in flight while the other stage's 8 WMMAs execute (partial s_wait_loadcnt).
template <bool F16OUT>
__global__ void gemm_wmma_kernel(const _Float16* __restrict__ A,
                                 const _Float16* __restrict__ B,
                                 void* __restrict__ C,
                                 int M, int N, int K,
                                 size_t strideB, size_t strideC) {
  const int b = blockIdx.z;
  const _Float16* Bb = B + (size_t)b * strideB;
  const int w  = threadIdx.x >> 5;
  const int wm = w & 3;
  const int wn = w >> 2;
  const int m0 = blockIdx.y * 128 + wm * 32;
  const int n0 = blockIdx.x * 128 + wn * 64;
  const int lane = threadIdx.x & 31;

  v8f acc[2][4] = {};
  v16h a[2][2], bf[2][4];

  // stage 0 preload (kk = 0)
  a[0][0] = load_frag(A, m0,      0, K);
  a[0][1] = load_frag(A, m0 + 16, 0, K);
#pragma unroll
  for (int j = 0; j < 4; ++j) bf[0][j] = load_frag(Bb, n0 + j * 16, 0, K);

  for (int kk = 0; kk < K; kk += 64) {
    // issue stage-1 loads (kk+32) before consuming stage 0
    if (kk + 32 < K) {
      a[1][0] = load_frag(A, m0,      kk + 32, K);
      a[1][1] = load_frag(A, m0 + 16, kk + 32, K);
#pragma unroll
      for (int j = 0; j < 4; ++j) bf[1][j] = load_frag(Bb, n0 + j * 16, kk + 32, K);
    }
    // prefetch two stages ahead into the cache hierarchy (global_prefetch_b8)
    if (kk + 128 < K) {
      __builtin_prefetch(A  + (size_t)(m0 + lane) * K + kk + 128, 0, 1);
      __builtin_prefetch(Bb + (size_t)(n0 + lane) * K + kk + 128, 0, 1);
      __builtin_prefetch(Bb + (size_t)(n0 + 32 + lane) * K + kk + 128, 0, 1);
    }
#pragma unroll
    for (int j = 0; j < 4; ++j) {
      acc[0][j] = wmma16(a[0][0], bf[0][j], acc[0][j]);
      acc[1][j] = wmma16(a[0][1], bf[0][j], acc[1][j]);
    }
    // refill stage 0 (kk+64) before consuming stage 1
    if (kk + 64 < K) {
      a[0][0] = load_frag(A, m0,      kk + 64, K);
      a[0][1] = load_frag(A, m0 + 16, kk + 64, K);
#pragma unroll
      for (int j = 0; j < 4; ++j) bf[0][j] = load_frag(Bb, n0 + j * 16, kk + 64, K);
    }
    if (kk + 32 < K) {
#pragma unroll
      for (int j = 0; j < 4; ++j) {
        acc[0][j] = wmma16(a[1][0], bf[1][j], acc[0][j]);
        acc[1][j] = wmma16(a[1][1], bf[1][j], acc[1][j]);
      }
    }
  }

#pragma unroll
  for (int i = 0; i < 2; ++i)
#pragma unroll
    for (int j = 0; j < 4; ++j) {
      if (F16OUT)
        store_tile_f16((_Float16*)C + (size_t)b * strideC, N,
                       m0 + i * 16, n0 + j * 16, acc[i][j]);
      else
        store_tile_f32((float*)C + (size_t)b * strideC, N,
                       m0 + i * 16, n0 + j * 16, acc[i][j]);
    }
}

// ---------------- attention softmaxes ----------------
// q: softmax over d (64) per (b,h,p); write qs[bh][p][d] * scale
__global__ void q_softmax_kernel(const _Float16* __restrict__ qkv,
                                 _Float16* __restrict__ qs) {
  int t = blockIdx.x * blockDim.x + threadIdx.x;   // 262144
  int p  = t & (P_ - 1);
  int bh = t >> 10;
  int b = bh >> 3, h = bh & 7;
  const _Float16* src = qkv + ((size_t)b * 1536 + h * DH_) * P_ + p;
  float vals[DH_];
  float mx = -3.4e38f;
#pragma unroll
  for (int d = 0; d < DH_; ++d) {
    float v = (float)src[(size_t)d * P_];
    vals[d] = v; mx = fmaxf(mx, v);
  }
  float s = 0.f;
#pragma unroll
  for (int d = 0; d < DH_; ++d) { float e = expf(vals[d] - mx); vals[d] = e; s += e; }
  float inv = 0.125f / s;   // * DIM_HEAD^-0.5
  _Float16* dst = qs + (size_t)bh * (P_ * DH_) + (size_t)p * DH_;
#pragma unroll
  for (int d = 0; d < DH_; ++d) dst[d] = (_Float16)(vals[d] * inv);
}

// k: softmax over n (4 mem + 1024), one wave per (b,h,d); write ks[bh*64+d][0..1055]
__global__ void k_softmax_kernel(const _Float16* __restrict__ qkv,
                                 const float* __restrict__ mem_kv,
                                 _Float16* __restrict__ ks) {
  int wave = blockIdx.x * 8 + (threadIdx.x >> 5);  // 16384 rows
  int lane = threadIdx.x & 31;
  int d  = wave & 63;
  int bh = wave >> 6;
  int b = bh >> 3, h = bh & 7;
  const _Float16* src = qkv + ((size_t)b * 1536 + 512 + h * DH_ + d) * P_;
  const float* mk = mem_kv + ((size_t)h * DH_ + d) * 4;   // mem_kv[0][h][d][*]
  float r[33];
  float mx = -3.4e38f;
#pragma unroll
  for (int i = 0; i < 33; ++i) {
    int n = lane + 32 * i;
    float v;
    if (n < 4)          v = mk[n];
    else if (n < NSEQ_) v = (float)src[n - 4];
    else                v = -3.4e38f;               // pad -> exp()==0
    r[i] = v; mx = fmaxf(mx, v);
  }
#pragma unroll
  for (int m = 16; m >= 1; m >>= 1) mx = fmaxf(mx, __shfl_xor(mx, m, 32));
  float s = 0.f;
#pragma unroll
  for (int i = 0; i < 33; ++i) { float e = expf(r[i] - mx); r[i] = e; s += e; }
#pragma unroll
  for (int m = 16; m >= 1; m >>= 1) s += __shfl_xor(s, m, 32);
  float inv = 1.f / s;
  _Float16* dst = ks + (size_t)wave * NPAD_ + lane;
#pragma unroll
  for (int i = 0; i < 33; ++i) dst[32 * i] = (_Float16)(r[i] * inv);
}

// v: concat mem_v, convert f16, zero pad -> vh[bh*64+e][0..1055]
__global__ void v_prep_kernel(const _Float16* __restrict__ qkv,
                              const float* __restrict__ mem_kv,
                              _Float16* __restrict__ vh, int total) {
  int t = blockIdx.x * blockDim.x + threadIdx.x;
  if (t >= total) return;
  int n   = t % NPAD_;
  int row = t / NPAD_;
  int e  = row & 63;
  int bh = row >> 6;
  int b = bh >> 3, h = bh & 7;
  float v;
  if (n < 4)
    v = mem_kv[(size_t)HEADS_ * DH_ * 4 + ((size_t)h * DH_ + e) * 4 + n];
  else if (n < NSEQ_)
    v = (float)qkv[((size_t)b * 1536 + 1024 + h * DH_ + e) * P_ + (n - 4)];
  else
    v = 0.f;
  vh[t] = (_Float16)v;
}

// context[d][e] = sum_n ks[d][n] * vh[e][n]; store transposed ctxT[bh][e][d]
__global__ void context_kernel(const _Float16* __restrict__ ks,
                               const _Float16* __restrict__ vh,
                               _Float16* __restrict__ ctxT) {
  const int w  = threadIdx.x >> 5;
  const int bh = blockIdx.x * 2 + (w >> 2);
  const int m0 = (w & 3) * 16;
  const _Float16* Ab = ks + (size_t)bh * DH_ * NPAD_;
  const _Float16* Bb = vh + (size_t)bh * DH_ * NPAD_;
  v8f acc[4] = {};
  v16h a[2], bf[2][4];

  a[0] = load_frag(Ab, m0, 0, NPAD_);
#pragma unroll
  for (int j = 0; j < 4; ++j) bf[0][j] = load_frag(Bb, j * 16, 0, NPAD_);

  for (int kk = 0; kk < NPAD_; kk += 64) {
    if (kk + 32 < NPAD_) {
      a[1] = load_frag(Ab, m0, kk + 32, NPAD_);
#pragma unroll
      for (int j = 0; j < 4; ++j) bf[1][j] = load_frag(Bb, j * 16, kk + 32, NPAD_);
    }
#pragma unroll
    for (int j = 0; j < 4; ++j) acc[j] = wmma16(a[0], bf[0][j], acc[j]);
    if (kk + 64 < NPAD_) {
      a[0] = load_frag(Ab, m0, kk + 64, NPAD_);
#pragma unroll
      for (int j = 0; j < 4; ++j) bf[0][j] = load_frag(Bb, j * 16, kk + 64, NPAD_);
    }
    if (kk + 32 < NPAD_) {
#pragma unroll
      for (int j = 0; j < 4; ++j) acc[j] = wmma16(a[1], bf[1][j], acc[j]);
    }
  }

  _Float16* base = ctxT + (size_t)bh * (DH_ * DH_);
#pragma unroll
  for (int j = 0; j < 4; ++j)
    store_tile_f16_T(base, DH_, m0, j * 16, acc[j]);
}

// out[e][p] = sum_d ctxT[e][d] * qs[p][d]; store out_attn[b][p][h*64+e]
__global__ void attn_out_kernel(const _Float16* __restrict__ ctxT,
                                const _Float16* __restrict__ qs,
                                _Float16* __restrict__ oa) {
  const int b = blockIdx.z, h = blockIdx.y;
  const int bh = b * HEADS_ + h;
  const int w  = threadIdx.x >> 5;
  const int n0 = blockIdx.x * 256 + w * 32;
  const _Float16* A  = ctxT + (size_t)bh * (DH_ * DH_);   // 64x64 (e x d)
  const _Float16* Bq = qs   + (size_t)bh * (P_ * DH_);    // 1024x64 (p x d)
  v8f acc[4][2] = {};
  v16h a[2][4], bf[2][2];

  // K = 64: issue every load first, then run all 16 WMMAs back-to-back
#pragma unroll
  for (int s = 0; s < 2; ++s) {
    const int kk = s * 32;
#pragma unroll
    for (int i = 0; i < 4; ++i) a[s][i] = load_frag(A, i * 16, kk, DH_);
    bf[s][0] = load_frag(Bq, n0,      kk, DH_);
    bf[s][1] = load_frag(Bq, n0 + 16, kk, DH_);
  }
#pragma unroll
  for (int s = 0; s < 2; ++s)
#pragma unroll
    for (int i = 0; i < 4; ++i) {
      acc[i][0] = wmma16(a[s][i], bf[s][0], acc[i][0]);
      acc[i][1] = wmma16(a[s][i], bf[s][1], acc[i][1]);
    }

  _Float16* base = oa + (size_t)b * P_ * HID_ + (size_t)h * DH_;
#pragma unroll
  for (int i = 0; i < 4; ++i)
#pragma unroll
    for (int j = 0; j < 2; ++j)
      store_tile_f16_T(base, HID_, i * 16, n0 + j * 16, acc[i][j]);
}

// final: out = rmsnorm(y + b_out) * g_out
__global__ void rmsnorm_out_kernel(const float* __restrict__ y,
                                   const float* __restrict__ bo,
                                   const float* __restrict__ g,
                                   float* __restrict__ out) {
  int t = blockIdx.x * blockDim.x + threadIdx.x;
  int b = t >> 10;
  int p = t & (P_ - 1);
  const float* yb = y + (size_t)b * C_ * P_ + p;
  float ss = 0.f;
  for (int c = 0; c < C_; ++c) { float v = yb[(size_t)c * P_] + bo[c]; ss += v * v; }
  float s = SQRT_C / fmaxf(sqrtf(ss), 1e-12f);
  float* ob = out + (size_t)b * C_ * P_ + p;
  for (int c = 0; c < C_; ++c)
    ob[(size_t)c * P_] = (yb[(size_t)c * P_] + bo[c]) * g[c] * s;
}

extern "C" void kernel_launch(void* const* d_in, const int* in_sizes, int n_in,
                              void* d_out, int out_size, void* d_ws, size_t ws_size,
                              hipStream_t stream) {
  const float* x      = (const float*)d_in[0];
  const float* g_in   = (const float*)d_in[1];
  const float* w_qkv  = (const float*)d_in[2];
  const float* mem_kv = (const float*)d_in[3];
  const float* w_out  = (const float*)d_in[4];
  const float* b_out  = (const float*)d_in[5];
  const float* g_out  = (const float*)d_in[6];
  float* out = (float*)d_out;
  (void)in_sizes; (void)n_in; (void)out_size; (void)ws_size;

  char* ws = (char*)d_ws;
  size_t off = 0;
  auto alloc = [&](size_t bytes) -> char* {
    char* p = ws + off;
    off += (bytes + 255) & ~(size_t)255;
    return p;
  };
  _Float16* xn    = (_Float16*)alloc((size_t)B_ * P_ * C_ * 2);           // 32 MB
  _Float16* wq16  = (_Float16*)alloc((size_t)3 * HID_ * C_ * 2);
  _Float16* wo16  = (_Float16*)alloc((size_t)C_ * HID_ * 2);
  _Float16* qkv16 = (_Float16*)alloc((size_t)B_ * 3 * HID_ * P_ * 2);     // 96 MB
  _Float16* qs    = (_Float16*)alloc((size_t)B_ * HEADS_ * P_ * DH_ * 2); // 32 MB
  _Float16* ks    = (_Float16*)alloc((size_t)B_ * HEADS_ * DH_ * NPAD_ * 2);
  _Float16* vh    = (_Float16*)alloc((size_t)B_ * HEADS_ * DH_ * NPAD_ * 2);
  _Float16* ctxT  = (_Float16*)alloc((size_t)B_ * HEADS_ * DH_ * DH_ * 2);
  _Float16* oa    = (_Float16*)alloc((size_t)B_ * P_ * HID_ * 2);         // 32 MB
  float*    y     = (float*)   alloc((size_t)B_ * C_ * P_ * 4);           // 64 MB

  convert_f16_kernel<<<(3 * HID_ * C_ + 255) / 256, 256, 0, stream>>>(w_qkv, wq16, 3 * HID_ * C_);
  convert_f16_kernel<<<(C_ * HID_ + 255) / 256, 256, 0, stream>>>(w_out, wo16, C_ * HID_);
  rmsnorm_in_kernel<<<(B_ * P_) / 256, 256, 0, stream>>>(x, g_in, xn);

  // qkv[b] (1536x1024) = wq16 (1536x512) * xn[b]^T
  gemm_wmma_kernel<true><<<dim3(P_ / 128, (3 * HID_) / 128, B_), 256, 0, stream>>>(
      wq16, xn, qkv16, 3 * HID_, P_, C_, (size_t)P_ * C_, (size_t)3 * HID_ * P_);

  q_softmax_kernel<<<(B_ * HEADS_ * P_) / 256, 256, 0, stream>>>(qkv16, qs);
  k_softmax_kernel<<<(B_ * HEADS_ * DH_) / 8, 256, 0, stream>>>(qkv16, mem_kv, ks);
  int vtotal = B_ * HEADS_ * DH_ * NPAD_;
  v_prep_kernel<<<(vtotal + 255) / 256, 256, 0, stream>>>(qkv16, mem_kv, vh, vtotal);

  context_kernel<<<(B_ * HEADS_) / 2, 256, 0, stream>>>(ks, vh, ctxT);
  attn_out_kernel<<<dim3(P_ / 256, HEADS_, B_), 256, 0, stream>>>(ctxT, qs, oa);

  // y[b] (512x1024) = wo16 (512x512) * oa[b]^T
  gemm_wmma_kernel<false><<<dim3(P_ / 128, C_ / 128, B_), 256, 0, stream>>>(
      wo16, oa, y, C_, P_, HID_, (size_t)P_ * HID_, (size_t)C_ * P_);

  rmsnorm_out_kernel<<<(B_ * P_) / 256, 256, 0, stream>>>(y, b_out, g_out, out);
}
